// ModernBertEncoderLayer_16776142258400
// MI455X (gfx1250) — compile-verified, matrix operand order checked by reading, not run on hardware
//
#include <hip/hip_runtime.h>
#include <hip/hip_bf16.h>

// ---------------- constants ----------------
#define S_LEN   2048
#define BATCH   2
#define HID     768
#define HEADS   12
#define HD      64
#define INTER   1152
#define QKV_N   (3 * HID)   // 2304
#define MROWS   (BATCH * S_LEN) // 4096
#define MASKV   (-10000.0f)

typedef __attribute__((ext_vector_type(16))) __bf16 v16bf;
typedef __attribute__((ext_vector_type(8)))  float  v8f;
typedef __attribute__((ext_vector_type(4)))  unsigned int u32x4;
typedef __attribute__((ext_vector_type(8)))  int i32x8;
typedef __attribute__((ext_vector_type(4)))  int i32x4;
typedef unsigned short ushort_t;

#if defined(__AMDGCN__) && __has_builtin(__builtin_amdgcn_tensor_load_to_lds) && \
    __has_builtin(__builtin_amdgcn_s_wait_tensorcnt)
#define HAVE_TDM 1
#else
#define HAVE_TDM 0
#endif

// f32 -> bf16, round-to-nearest (ties away): cheap (1 add + shift)
__device__ __forceinline__ ushort_t f2bf(float f) {
    return (ushort_t)((__float_as_uint(f) + 0x8000u) >> 16);
}
// pack two f32 -> packed bf16x2 via v_perm_b32
__device__ __forceinline__ unsigned int pack2(float a, float b) {
    unsigned int ua = __float_as_uint(a) + 0x8000u;
    unsigned int ub = __float_as_uint(b) + 0x8000u;
    return __builtin_amdgcn_perm(ub, ua, 0x07060302u);
}

// Load one 16x32 bf16 A/B fragment for this lane from an LDS row holding the
// lane's 32 K-values contiguously (16B aligned): element e maps to
// k = ((e&8)<<1) + (lane>=16 ? 8 : 0) + (e&7) -> two ds_load_b128.
__device__ __forceinline__ v16bf load_frag_row(const ushort_t* row, int lane) {
    int base = (lane & 16) >> 1;   // 0 or 8 ushorts
    union { uint4 q[2]; v16bf v; } u;
    u.q[0] = *(const uint4*)(row + base);
    u.q[1] = *(const uint4*)(row + 16 + base);
    return u.v;
}

__device__ __forceinline__ v8f wmma_bf16(v16bf a, v16bf b, v8f c) {
    return __builtin_amdgcn_wmma_f32_16x16x32_bf16(
        false, a, false, b, (short)0, c, false, false);
}

#if HAVE_TDM
// TDM: DMA a 64-row x 32-bf16 tile (row stride K elements) into LDS rows
// padded to 80B (64B data + 16B pad -> pad_interval=16 dwords, pad_amount=4).
__device__ __forceinline__ void tdm_load_b_tile(const ushort_t* gsrc,
                                                unsigned int lds_base, int K) {
    unsigned long long ga = (unsigned long long)(size_t)gsrc;
    u32x4 g0;
    g0[0] = 1u;                                   // count=1 (valid D#)
    g0[1] = lds_base;                             // lds_addr
    g0[2] = (unsigned int)(ga & 0xFFFFFFFFull);   // global_addr[31:0]
    g0[3] = (unsigned int)((ga >> 32) & 0x1FFFFFFull) |
            (2u << 30);                           // ga[56:32] | type=2
    i32x8 g1;
    g1[0] = (int)((1u << 16)      // data_size = 2 bytes
                | (1u << 20)      // pad_enable
                | (3u << 22)      // pad_interval: 16 dwords (64B)
                | (3u << 25));    // pad_amount: 4 dwords (16B)
    g1[1] = (int)(32u << 16);     // tensor_dim0 = 32 elements
    g1[2] = (int)(64u << 16);     // tensor_dim1 = 64 rows
    g1[3] = (int)(32u << 16);     // tile_dim0 = 32 elements
    g1[4] = 64;                   // tile_dim1 = 64 rows
    g1[5] = K;                    // tensor_dim0_stride (elements)
    g1[6] = 0; g1[7] = 0;
    i32x4 z4 = {0, 0, 0, 0};
    i32x8 z8 = {0, 0, 0, 0, 0, 0, 0, 0};
    __builtin_amdgcn_tensor_load_to_lds(g0, g1, z4, z4, z8, 0);
}
#endif

// ------------- weight convert + transpose: in[K][N] f32 -> out[N][K] bf16 ----
__global__ void convT_bf16_kernel(const float* __restrict__ in,
                                  ushort_t* __restrict__ out, int K, int N) {
    int i = blockIdx.x * blockDim.x + threadIdx.x;
    if (i < N * K) {
        int n = i / K;
        int k = i - n * K;
        out[i] = f2bf(in[(size_t)k * N + n]);
    }
}

// ---------------- layernorm (row = 768) ----------------
__global__ __launch_bounds__(256)
void layernorm_kernel(const float* __restrict__ x,
                      const float* __restrict__ g,
                      const float* __restrict__ bta,
                      float* __restrict__ out) {
    __shared__ float red1[256];
    __shared__ float red2[256];
    int row = blockIdx.x;
    int tid = threadIdx.x;
    const float* xr = x + (size_t)row * HID;
    float v[3]; float s = 0.f, ss = 0.f;
#pragma unroll
    for (int j = 0; j < 3; ++j) {
        v[j] = xr[tid + j * 256];
        s += v[j]; ss += v[j] * v[j];
    }
    red1[tid] = s; red2[tid] = ss;
    __syncthreads();
    for (int off = 128; off > 0; off >>= 1) {
        if (tid < off) { red1[tid] += red1[tid + off]; red2[tid] += red2[tid + off]; }
        __syncthreads();
    }
    float mu   = red1[0] * (1.0f / HID);
    float var  = red2[0] * (1.0f / HID) - mu * mu;
    float rstd = rsqrtf(var + 1e-5f);
    float* orow = out + (size_t)row * HID;
#pragma unroll
    for (int j = 0; j < 3; ++j) {
        int c = tid + j * 256;
        orow[c] = (v[j] - mu) * rstd * g[c] + bta[c];
    }
}

// ---------------- GeGLU: out = gelu_tanh(h0) * h1 ----------------
__global__ void geglu_kernel(const float* __restrict__ h0,
                             const float* __restrict__ h1,
                             float* __restrict__ out, int n) {
    int i = blockIdx.x * blockDim.x + threadIdx.x;
    if (i < n) {
        float x = h0[i];
        float t = tanhf(0.7978845608028654f * (x + 0.044715f * x * x * x));
        out[i] = 0.5f * x * (1.0f + t) * h1[i];
    }
}

// ------- GEMM: C[M,N] = A_f32[M,K] @ W[N][K](bf16, pre-transposed) (+res) ----
// 256 threads (8 waves), 64x64 C tile, K chunks of 32 staged in LDS.
// B tiles are double-buffered and DMA'd by the Tensor Data Mover: wave 0
// issues the TDM for chunk i+1 before computing chunk i, and only waits
// s_wait_tensorcnt at the end-of-iteration barrier -> DMA overlaps WMMA.
__global__ __launch_bounds__(256)
void gemm_f32a_bf16bt(const float* __restrict__ A,
                      const ushort_t* __restrict__ Bt,  // [N][K] bf16
                      const float* __restrict__ residual,
                      float* __restrict__ C,
                      int M, int N, int K) {
    __shared__ __align__(16) ushort_t lA[64][40];       // [m][k], stride 80B
    __shared__ __align__(16) ushort_t lBt[2][64][40];   // [buf][n][k]
    int tid  = threadIdx.x;
    int lane = tid & 31;
    int wave = tid >> 5;
    int n0 = blockIdx.x * 64;
    int m0 = blockIdx.y * 64;
    int sub0 = wave * 2;
    int ti  = sub0 >> 2;        // row tile 0..3 (same for both sub-tiles)
    int tj0 = sub0 & 3;         // col tiles tj0, tj0+1
    v8f acc0 = {}; v8f acc1 = {};

    int ra = tid >> 2;              // A row 0..63
    int ca = (tid & 3) * 8;         // A col group {0,8,16,24}

#if HAVE_TDM
    // prologue: synchronously DMA chunk 0 into buffer 0
    if (wave == 0) {
        tdm_load_b_tile(Bt + (size_t)n0 * K, (unsigned int)(size_t)&lBt[0][0][0], K);
        __builtin_amdgcn_s_wait_tensorcnt(0);
    }
#endif

    int buf = 0;
    for (int kb = 0; kb < K; kb += 32, buf ^= 1) {
        // ---- stage A: two float4 loads -> 8 bf16 -> one ds_store_b128 ----
        const float* ap = A + (size_t)(m0 + ra) * K + kb + ca;
        float4 fa = *(const float4*)(ap);
        float4 fb = *(const float4*)(ap + 4);
        uint4 pa;
        pa.x = pack2(fa.x, fa.y); pa.y = pack2(fa.z, fa.w);
        pa.z = pack2(fb.x, fb.y); pa.w = pack2(fb.z, fb.w);
        *(uint4*)&lA[ra][ca] = pa;
        if (kb + 32 < K) __builtin_prefetch(ap + 32, 0, 0);

        // ---- B tile for chunk kb is already in lBt[buf] ----
#if HAVE_TDM
        // async-issue DMA of next chunk into the other buffer
        if (wave == 0 && kb + 32 < K) {
            tdm_load_b_tile(Bt + (size_t)n0 * K + kb + 32,
                            (unsigned int)(size_t)&lBt[buf ^ 1][0][0], K);
        }
#else
        {
            int rb = tid >> 2;              // Bt row (n) 0..63
            int cb = (tid & 3) * 8;         // col group {0,8,16,24}
            *(uint4*)&lBt[buf][rb][cb] =
                *(const uint4*)(Bt + (size_t)(n0 + rb) * K + kb + cb);
        }
#endif
        __syncthreads();   // publish lA (and, first iter, lBt[0])

        v16bf a  = load_frag_row(&lA[ti * 16 + (lane & 15)][0], lane);
        v16bf b0 = load_frag_row(&lBt[buf][tj0 * 16 + (lane & 15)][0], lane);
        v16bf b1 = load_frag_row(&lBt[buf][tj0 * 16 + 16 + (lane & 15)][0], lane);
        acc0 = wmma_bf16(a, b0, acc0);
        acc1 = wmma_bf16(a, b1, acc1);

#if HAVE_TDM
        // next buffer's DMA must be complete before the barrier releases it
        if (wave == 0) __builtin_amdgcn_s_wait_tensorcnt(0);
#endif
        __syncthreads();
    }
#pragma unroll
    for (int r = 0; r < 8; ++r) {
        int mm  = m0 + ti * 16 + r + ((lane >> 4) << 3);
        int nn0 = n0 + tj0 * 16 + (lane & 15);
        size_t i0 = (size_t)mm * N + nn0;
        size_t i1 = i0 + 16;
        float v0 = acc0[r], v1 = acc1[r];
        if (residual) { v0 += residual[i0]; v1 += residual[i1]; }
        C[i0] = v0; C[i1] = v1;
    }
}

// ---------------- sliding-window attention ----------------
// 1 wave per (q-tile of 16, head, batch). WMMA for QK^T and P@V.
__global__ __launch_bounds__(32)
void attn_kernel(const float* __restrict__ qkv,
                 const int* __restrict__ pad,
                 float* __restrict__ ctx) {
    __shared__ __align__(16) ushort_t sQ[16][64];   // [q][feat]
    __shared__ __align__(16) ushort_t sK[16][64];   // [key][feat]
    __shared__ __align__(16) ushort_t sVt[64][40];  // [feat][key-in-chunk]
    __shared__ __align__(16) float    sS[16][160];
    __shared__ __align__(16) ushort_t sP[16][160];

    int lane = threadIdx.x;
    int qt = blockIdx.x, h = blockIdx.y, b = blockIdx.z;
    int q0 = qt * 16;
    size_t rowbase = (size_t)b * S_LEN;

    // ---- stage Q (pre-scaled by 1/sqrt(64) = 0.125) ----
#pragma unroll
    for (int i = 0; i < 8; ++i) {
        int off = (lane + i * 32) * 4;
        int r = off >> 6, c = off & 63;
        float4 f = *(const float4*)(qkv + (rowbase + q0 + r) * QKV_N + h * HD + c);
        uint2 p;
        p.x = pack2(f.x * 0.125f, f.y * 0.125f);
        p.y = pack2(f.z * 0.125f, f.w * 0.125f);
        *(uint2*)&sQ[r][c] = p;
    }
    __syncthreads();
    v16bf aQ0 = load_frag_row(&sQ[lane & 15][0], lane);
    v16bf aQ1 = load_frag_row(&sQ[lane & 15][32], lane);

    int t_lo = qt - 4; if (t_lo < 0) t_lo = 0;
    int t_hi = qt + 4; if (t_hi > (S_LEN / 16 - 1)) t_hi = S_LEN / 16 - 1;
    int nt = t_hi - t_lo + 1;             // 5..9 k-tiles

    // ---- scores: Q @ K^T per k-tile ----
    for (int tt = 0; tt < nt; ++tt) {
        int kt = t_lo + tt;
#pragma unroll
        for (int i = 0; i < 8; ++i) {
            int off = (lane + i * 32) * 4;
            int r = off >> 6, c = off & 63;
            float4 f = *(const float4*)(qkv + (rowbase + kt * 16 + r) * QKV_N +
                                        HID + h * HD + c);
            uint2 p; p.x = pack2(f.x, f.y); p.y = pack2(f.z, f.w);
            *(uint2*)&sK[r][c] = p;
        }
        __syncthreads();
        v16bf b0 = load_frag_row(&sK[lane & 15][0], lane);
        v16bf b1 = load_frag_row(&sK[lane & 15][32], lane);
        v8f acc = {};
        acc = wmma_bf16(aQ0, b0, acc);
        acc = wmma_bf16(aQ1, b1, acc);
#pragma unroll
        for (int r = 0; r < 8; ++r) {
            int m = r + ((lane >> 4) << 3);
            int n = lane & 15;
            int qi = q0 + m;
            int ki = kt * 16 + n;
            float v = acc[r];
            int d = qi - ki; if (d < 0) d = -d;
            if (d > 64) v += MASKV;                           // sliding window
            v += (1.0f - (float)pad[b * S_LEN + ki]) * MASKV; // padding mask
            sS[m][tt * 16 + n] = v;
        }
        __syncthreads();
    }

    // ---- softmax (lanes 0..15, one row each) ----
    int ncols   = nt * 16;
    int nchunks = (nt + 1) >> 1;
    int ptot    = nchunks * 32;
    if (lane < 16) {
        float mx = -3.0e38f;
        for (int c = 0; c < ncols; ++c) mx = fmaxf(mx, sS[lane][c]);
        float sum = 0.f;
        for (int c = 0; c < ncols; ++c) {
            float e = __expf(sS[lane][c] - mx);
            sS[lane][c] = e; sum += e;
        }
        float inv = 1.0f / sum;
        for (int c = 0; c < ncols; ++c) sP[lane][c] = f2bf(sS[lane][c] * inv);
        for (int c = ncols; c < ptot; ++c) sP[lane][c] = 0;
    }
    __syncthreads();

    // ---- P @ V over 32-key chunks ----
    v8f o0 = {}, o1 = {}, o2 = {}, o3 = {};
    for (int ch = 0; ch < nchunks; ++ch) {
        // stage V transposed: [feat][key]
#pragma unroll
        for (int i = 0; i < 16; ++i) {
            int off = (lane + i * 32) * 4;
            int r = off >> 6;          // key within chunk 0..31
            int c = off & 63;          // feat group
            int gk = t_lo * 16 + ch * 32 + r;
            if (gk > S_LEN - 1) gk = S_LEN - 1;   // padded P is zero there
            float4 f = *(const float4*)(qkv + (rowbase + gk) * QKV_N +
                                        2 * HID + h * HD + c);
            sVt[c + 0][r] = f2bf(f.x);
            sVt[c + 1][r] = f2bf(f.y);
            sVt[c + 2][r] = f2bf(f.z);
            sVt[c + 3][r] = f2bf(f.w);
        }
        __syncthreads();
        v16bf aP = load_frag_row(&sP[lane & 15][0] + ch * 32, lane);
        v16bf bv;
        bv = load_frag_row(&sVt[(lane & 15)][0], lane);      o0 = wmma_bf16(aP, bv, o0);
        bv = load_frag_row(&sVt[16 + (lane & 15)][0], lane); o1 = wmma_bf16(aP, bv, o1);
        bv = load_frag_row(&sVt[32 + (lane & 15)][0], lane); o2 = wmma_bf16(aP, bv, o2);
        bv = load_frag_row(&sVt[48 + (lane & 15)][0], lane); o3 = wmma_bf16(aP, bv, o3);
        __syncthreads();
    }

    // ---- store context (row-major [4096, 768], head-major cols) ----
#pragma unroll
    for (int r = 0; r < 8; ++r) {
        int m = r + ((lane >> 4) << 3);
        int n = lane & 15;
        size_t base = (rowbase + q0 + m) * HID + h * HD;
        ctx[base + n]      = o0[r];
        ctx[base + 16 + n] = o1[r];
        ctx[base + 32 + n] = o2[r];
        ctx[base + 48 + n] = o3[r];
    }
}

// ---------------- host launch ----------------
extern "C" void kernel_launch(void* const* d_in, const int* in_sizes, int n_in,
                              void* d_out, int out_size, void* d_ws, size_t ws_size,
                              hipStream_t stream) {
    const float* x        = (const float*)d_in[0];
    const float* g1       = (const float*)d_in[1];
    const float* b1       = (const float*)d_in[2];
    const float* Wqkv     = (const float*)d_in[3];
    const float* Wo       = (const float*)d_in[4];
    const float* g2       = (const float*)d_in[5];
    const float* b2       = (const float*)d_in[6];
    const float* wi0      = (const float*)d_in[7];
    const float* wi1      = (const float*)d_in[8];
    const float* wo       = (const float*)d_in[9];
    const int*   padmask  = (const int*)d_in[10];
    float* out = (float*)d_out;

    // ---- workspace layout ----
    char* p = (char*)d_ws;
    float* attn_res = (float*)p;                 p += (size_t)MROWS * HID * 4;
    float* qkv      = (float*)p;                 p += (size_t)MROWS * QKV_N * 4;
    float* context  = (float*)p;                 p += (size_t)MROWS * HID * 4;
    float* x1       = (float*)p;                 p += (size_t)MROWS * HID * 4;
    float* mlp_res  = (float*)p;                 p += (size_t)MROWS * HID * 4;
    float* gated    = (float*)p;                 p += (size_t)MROWS * INTER * 4;
    ushort_t* wqkv_bf = (ushort_t*)p;            p += (size_t)HID * QKV_N * 2;
    ushort_t* wo_bf   = (ushort_t*)p;            p += (size_t)HID * HID * 2;
    ushort_t* wi0_bf  = (ushort_t*)p;            p += (size_t)HID * INTER * 2;
    ushort_t* wi1_bf  = (ushort_t*)p;            p += (size_t)HID * INTER * 2;
    ushort_t* wom_bf  = (ushort_t*)p;            p += (size_t)INTER * HID * 2;
    // h0/h1 alias the (dead by then) qkv buffer
    float* h0 = qkv;
    float* h1 = qkv + (size_t)MROWS * INTER;

    // ---- weight conversion f32[K][N] -> bf16[N][K] (transposed) ----
    auto convT = [&](const float* src, ushort_t* dst, int K, int N) {
        int n = K * N;
        convT_bf16_kernel<<<(n + 255) / 256, 256, 0, stream>>>(src, dst, K, N);
    };
    convT(Wqkv, wqkv_bf, HID, QKV_N);
    convT(Wo,   wo_bf,   HID, HID);
    convT(wi0,  wi0_bf,  HID, INTER);
    convT(wi1,  wi1_bf,  HID, INTER);
    convT(wo,   wom_bf,  INTER, HID);

    // 1. attn_res = LN(x)
    layernorm_kernel<<<MROWS, 256, 0, stream>>>(x, g1, b1, attn_res);

    // 2. qkv = attn_res @ Wqkv
    gemm_f32a_bf16bt<<<dim3(QKV_N / 64, MROWS / 64), 256, 0, stream>>>(
        attn_res, wqkv_bf, nullptr, qkv, MROWS, QKV_N, HID);

    // 3. sliding-window attention -> context
    attn_kernel<<<dim3(S_LEN / 16, HEADS, BATCH), 32, 0, stream>>>(
        qkv, padmask, context);

    // 4. x1 = attn_res + context @ Wo
    gemm_f32a_bf16bt<<<dim3(HID / 64, MROWS / 64), 256, 0, stream>>>(
        context, wo_bf, attn_res, x1, MROWS, HID, HID);

    // 5. mlp_res = LN(x1)
    layernorm_kernel<<<MROWS, 256, 0, stream>>>(x1, g2, b2, mlp_res);

    // 6/7. h0 = mlp_res @ wi0 ; h1 = mlp_res @ wi1
    gemm_f32a_bf16bt<<<dim3(INTER / 64, MROWS / 64), 256, 0, stream>>>(
        mlp_res, wi0_bf, nullptr, h0, MROWS, INTER, HID);
    gemm_f32a_bf16bt<<<dim3(INTER / 64, MROWS / 64), 256, 0, stream>>>(
        mlp_res, wi1_bf, nullptr, h1, MROWS, INTER, HID);

    // 8. gated = gelu(h0) * h1
    {
        int n = MROWS * INTER;
        geglu_kernel<<<(n + 255) / 256, 256, 0, stream>>>(h0, h1, gated, n);
    }

    // 9. out = mlp_res + gated @ wo
    gemm_f32a_bf16bt<<<dim3(HID / 64, MROWS / 64), 256, 0, stream>>>(
        gated, wom_bf, mlp_res, out, MROWS, HID, INTER);
}
